// SCHNET_25340307046986
// MI455X (gfx1250) — compile-verified, matrix operand order checked by reading, not run on hardware
//
#include <hip/hip_runtime.h>
#include <hip/hip_bf16.h>
#include <math.h>

typedef __attribute__((ext_vector_type(16))) _Float16 v16h;
typedef __attribute__((ext_vector_type(8)))  _Float16 v8h;
typedef __attribute__((ext_vector_type(4)))  _Float16 v4h;
typedef __attribute__((ext_vector_type(8)))  float    v8f;

#define HDIM   128
#define GCOUNT 50
#define CUTOFF_F 10.0f
#define PI_F 3.14159265358979323846f

union V16U { v16h v; v8h h[2]; };

__device__ __forceinline__ v8f wmma16(v16h a, v16h b, v8f c) {
  // D = A(16x32 f16) x B(32x16 f16) + C(16x16 f32)
  return __builtin_amdgcn_wmma_f32_16x16x32_f16(false, a, false, b, (short)0, c, false, false);
}

__device__ __forceinline__ float sspf(float v) {
  // softplus(v) - log(2), numerically stable
  return fmaxf(v, 0.0f) + log1pf(__expf(-fabsf(v))) - 0.69314718055994531f;
}

// B operand, pre-swizzled in workspace: [tile][lane][16 halves] (16B x2 per lane)
__device__ __forceinline__ v16h load_b_swz(const _Float16* __restrict__ ws, int tile, int lane) {
  V16U u;
  const v8h* p = (const v8h*)(ws + ((size_t)(tile * 32 + lane) << 4));
  u.h[0] = p[0];
  u.h[1] = p[1];
  return u.v;
}

// A operand from LDS tile buffer a1[16][128] f16, k-step s (32 k's per step).
__device__ __forceinline__ v16h load_a_lds(const _Float16* a1, int lo, int hi, int s) {
  V16U u;
  const _Float16* p = a1 + lo * HDIM + 8 * hi + 32 * s;
  u.h[0] = *(const v8h*)p;
  u.h[1] = *(const v8h*)(p + 16);
  return u.v;
}

// stage one 16x128 f32 row-tile into LDS as f16 (b128 loads, b64 stores)
__device__ __forceinline__ void stage_tile(const float* __restrict__ src,
                                           _Float16* __restrict__ a1, int lane) {
  const float4* xp = (const float4*)src;
  v4h* dp = (v4h*)a1;
  for (int idx = lane; idx < (16 * HDIM) / 4; idx += 32) {
    float4 v = xp[idx];
    v4h p;
    p[0] = (_Float16)v.x; p[1] = (_Float16)v.y;
    p[2] = (_Float16)v.z; p[3] = (_Float16)v.w;
    dp[idx] = p;
  }
}

// ---------------- weight pre-swizzle into WMMA B layout -------------------
__global__ void swizzle_weight(const float* __restrict__ W, _Float16* __restrict__ dst,
                               int K, int nTiles) {
  int total = nTiles * 512;
  for (int idx = blockIdx.x * blockDim.x + threadIdx.x; idx < total;
       idx += gridDim.x * blockDim.x) {
    int tile = idx >> 9;
    int lane = (idx >> 4) & 31;
    int i    = idx & 15;
    int lo = lane & 15, hi = lane >> 4;
    int k = ((i < 8) ? (8 * hi + i) : (8 + 8 * hi + i)) + 32 * (tile >> 3);
    int n = lo + 16 * (tile & 7);
    float v = (k < K) ? W[(size_t)k * HDIM + n] : 0.0f;
    dst[idx] = (_Float16)v;
  }
}

// ---------------- edge prep: distance + cosine cutoff ---------------------
__global__ void prep_edges(const float* __restrict__ pos, const int* __restrict__ ei,
                           float* __restrict__ dbuf, float* __restrict__ ccut, int E) {
  for (int e = blockIdx.x * blockDim.x + threadIdx.x; e < E;
       e += gridDim.x * blockDim.x) {
    unsigned r = (unsigned)ei[e], c = (unsigned)ei[E + e];
    float dx = pos[3 * r]     - pos[3 * c];
    float dy = pos[3 * r + 1] - pos[3 * c + 1];
    float dz = pos[3 * r + 2] - pos[3 * c + 2];
    float d = sqrtf(dx * dx + dy * dy + dz * dz + 1e-12f);
    dbuf[e] = d;
    ccut[e] = 0.5f * (__cosf(d * (PI_F / CUTOFF_F)) + 1.0f);
  }
}

__global__ void zerof(float* __restrict__ p, int n) {
  for (int i = blockIdx.x * blockDim.x + threadIdx.x; i < n;
       i += gridDim.x * blockDim.x) p[i] = 0.0f;
}

// ---------------- h = x @ cf_lin1_w  (WMMA) -------------------------------
__global__ void __launch_bounds__(256) node_gemm(const float* __restrict__ xin,
                                                 const _Float16* __restrict__ wsz,
                                                 float* __restrict__ out, int N) {
  __shared__ _Float16 ax[8][16 * HDIM];
  int wave = threadIdx.x >> 5, lane = threadIdx.x & 31;
  int lo = lane & 15, hi = lane >> 4;
  int gw = blockIdx.x * 8 + wave, nw = gridDim.x * 8;
  int ntiles = N >> 4;
  _Float16* a1 = ax[wave];
  for (int t = gw; t < ntiles; t += nw) {
    int n0 = t << 4;
    stage_tile(xin + (size_t)n0 * HDIM, a1, lane);
    __builtin_amdgcn_wave_barrier();
    v8f acc[8];
#pragma unroll
    for (int c = 0; c < 8; ++c)
#pragma unroll
      for (int j = 0; j < 8; ++j) acc[c][j] = 0.0f;
#pragma unroll
    for (int s = 0; s < 4; ++s) {
      v16h a = load_a_lds(a1, lo, hi, s);
#pragma unroll
      for (int c = 0; c < 8; ++c)
        acc[c] = wmma16(a, load_b_swz(wsz, s * 8 + c, lane), acc[c]);
    }
    // single base pointer; all 64 stores use compile-time immediate offsets
    float* op = out + (size_t)(unsigned)(n0 + 8 * hi) * HDIM + lo;
#pragma unroll
    for (int j = 0; j < 8; ++j)
#pragma unroll
      for (int c = 0; c < 8; ++c) op[j * HDIM + 16 * c] = acc[c][j];
    __builtin_amdgcn_wave_barrier();
  }
}

// ---- fused edge kernel: gaussians -> MLP (2x WMMA GEMM) -> gather/scatter
__global__ void __launch_bounds__(256) edge_msg(
    const float* __restrict__ dbuf, const float* __restrict__ ccut,
    const int* __restrict__ ei, const float* __restrict__ hbuf,
    const _Float16* __restrict__ w1sz, const _Float16* __restrict__ w2sz,
    const float* __restrict__ b1, const float* __restrict__ b2,
    float* __restrict__ agg, int E) {
  __shared__ _Float16 ax[8][16 * HDIM];
  int wave = threadIdx.x >> 5, lane = threadIdx.x & 31;
  int lo = lane & 15, hi = lane >> 4;
  const float delta = CUTOFF_F / (float)(GCOUNT - 1);
  const float coeff = -0.5f / (delta * delta);
  float bias1[8], bias2[8];
#pragma unroll
  for (int c = 0; c < 8; ++c) {
    bias1[c] = b1[lo + 16 * c];
    bias2[c] = b2[lo + 16 * c];
  }
  int gw = blockIdx.x * 8 + wave, nw = gridDim.x * 8;
  int ntiles = E >> 4;
  _Float16* a1 = ax[wave];
  for (int t = gw; t < ntiles; t += nw) {
    int e0 = t << 4;
    float dv = dbuf[e0 + lo];  // edge row this lane represents in A layout
    // GEMM1: edge_attr(16x64 gaussians, padded) @ nn_w1(64x128) + b1
    v8f acc[8];
#pragma unroll
    for (int c = 0; c < 8; ++c)
#pragma unroll
      for (int j = 0; j < 8; ++j) acc[c][j] = bias1[c];
#pragma unroll
    for (int s = 0; s < 2; ++s) {
      v16h a;
#pragma unroll
      for (int i = 0; i < 16; ++i) {
        int k = ((i < 8) ? (8 * hi + i) : (8 + 8 * hi + i)) + 32 * s;
        float tt = dv - (float)k * delta;
        float g = (k < GCOUNT) ? __expf(coeff * tt * tt) : 0.0f;
        a[i] = (_Float16)g;
      }
#pragma unroll
      for (int c = 0; c < 8; ++c)
        acc[c] = wmma16(a, load_b_swz(w1sz, s * 8 + c, lane), acc[c]);
    }
    // ssp -> repack (C layout -> A layout) through per-wave LDS
    {
      _Float16* sp = a1 + (8 * hi) * HDIM + lo;
#pragma unroll
      for (int j = 0; j < 8; ++j)
#pragma unroll
        for (int c = 0; c < 8; ++c)
          sp[j * HDIM + 16 * c] = (_Float16)sspf(acc[c][j]);
    }
    __builtin_amdgcn_wave_barrier();
    // GEMM2: ssp_out(16x128) @ nn_w2(128x128) + b2
    v8f acc2[8];
#pragma unroll
    for (int c = 0; c < 8; ++c)
#pragma unroll
      for (int j = 0; j < 8; ++j) acc2[c][j] = bias2[c];
#pragma unroll
    for (int s = 0; s < 4; ++s) {
      v16h a = load_a_lds(a1, lo, hi, s);
#pragma unroll
      for (int c = 0; c < 8; ++c)
        acc2[c] = wmma16(a, load_b_swz(w2sz, s * 8 + c, lane), acc2[c]);
    }
    // epilogue: Wf *= C; m = h[row]*Wf; atomic scatter into agg[col]
    int4 r0 = *(const int4*)(ei + e0 + 8 * hi);
    int4 r1 = *(const int4*)(ei + e0 + 8 * hi + 4);
    int4 c0 = *(const int4*)(ei + E + e0 + 8 * hi);
    int4 c1 = *(const int4*)(ei + E + e0 + 8 * hi + 4);
    float4 f0 = *(const float4*)(ccut + e0 + 8 * hi);
    float4 f1 = *(const float4*)(ccut + e0 + 8 * hi + 4);
    unsigned re[8] = {(unsigned)r0.x, (unsigned)r0.y, (unsigned)r0.z, (unsigned)r0.w,
                      (unsigned)r1.x, (unsigned)r1.y, (unsigned)r1.z, (unsigned)r1.w};
    unsigned ce[8] = {(unsigned)c0.x, (unsigned)c0.y, (unsigned)c0.z, (unsigned)c0.w,
                      (unsigned)c1.x, (unsigned)c1.y, (unsigned)c1.z, (unsigned)c1.w};
    float    cf[8] = {f0.x, f0.y, f0.z, f0.w, f1.x, f1.y, f1.z, f1.w};
#pragma unroll
    for (int j = 0; j < 8; ++j) {
      const float* hp = hbuf + ((size_t)re[j] << 7) + lo;  // row gather base
      float*       ap = agg  + ((size_t)ce[j] << 7) + lo;  // col scatter base
#pragma unroll
      for (int c = 0; c < 8; ++c)
        atomicAdd(ap + 16 * c, acc2[c][j] * cf[j] * hp[16 * c]);
    }
    __builtin_amdgcn_wave_barrier();
  }
}

// -- fused node MLP: x = ssp(agg @ cf_lin2 + b) @ post_w + post_b (2x WMMA)
__global__ void __launch_bounds__(256) node_mlp2(
    const float* __restrict__ agg,
    const _Float16* __restrict__ w2sz, const float* __restrict__ b2,
    const _Float16* __restrict__ w3sz, const float* __restrict__ b3,
    float* __restrict__ xout, int N) {
  __shared__ _Float16 ax[8][16 * HDIM];
  int wave = threadIdx.x >> 5, lane = threadIdx.x & 31;
  int lo = lane & 15, hi = lane >> 4;
  float bias2[8], bias3[8];
#pragma unroll
  for (int c = 0; c < 8; ++c) {
    bias2[c] = b2[lo + 16 * c];
    bias3[c] = b3[lo + 16 * c];
  }
  int gw = blockIdx.x * 8 + wave, nw = gridDim.x * 8;
  int ntiles = N >> 4;
  _Float16* a1 = ax[wave];
  for (int t = gw; t < ntiles; t += nw) {
    int n0 = t << 4;
    stage_tile(agg + (size_t)n0 * HDIM, a1, lane);
    __builtin_amdgcn_wave_barrier();
    v8f acc[8];
#pragma unroll
    for (int c = 0; c < 8; ++c)
#pragma unroll
      for (int j = 0; j < 8; ++j) acc[c][j] = bias2[c];
#pragma unroll
    for (int s = 0; s < 4; ++s) {
      v16h a = load_a_lds(a1, lo, hi, s);
#pragma unroll
      for (int c = 0; c < 8; ++c)
        acc[c] = wmma16(a, load_b_swz(w2sz, s * 8 + c, lane), acc[c]);
    }
    __builtin_amdgcn_wave_barrier();
    {
      _Float16* sp = a1 + (8 * hi) * HDIM + lo;
#pragma unroll
      for (int j = 0; j < 8; ++j)
#pragma unroll
        for (int c = 0; c < 8; ++c)
          sp[j * HDIM + 16 * c] = (_Float16)sspf(acc[c][j]);
    }
    __builtin_amdgcn_wave_barrier();
    v8f acc2[8];
#pragma unroll
    for (int c = 0; c < 8; ++c)
#pragma unroll
      for (int j = 0; j < 8; ++j) acc2[c][j] = bias3[c];
#pragma unroll
    for (int s = 0; s < 4; ++s) {
      v16h a = load_a_lds(a1, lo, hi, s);
#pragma unroll
      for (int c = 0; c < 8; ++c)
        acc2[c] = wmma16(a, load_b_swz(w3sz, s * 8 + c, lane), acc2[c]);
    }
    float* op = xout + (size_t)(unsigned)(n0 + 8 * hi) * HDIM + lo;
#pragma unroll
    for (int j = 0; j < 8; ++j)
#pragma unroll
      for (int c = 0; c < 8; ++c) op[j * HDIM + 16 * c] = acc2[c][j];
    __builtin_amdgcn_wave_barrier();
  }
}

// ---------------- head: ssp(x@W1+b1)@W2+b2 -> per-graph atomic sum --------
__global__ void head_reduce(const float* __restrict__ xin,
                            const float* __restrict__ w1, const float* __restrict__ b1,
                            const float* __restrict__ w2, const float* __restrict__ b2,
                            const int* __restrict__ batch, float* __restrict__ out, int N) {
  for (int n = blockIdx.x * blockDim.x + threadIdx.x; n < N;
       n += gridDim.x * blockDim.x) {
    const float* xr = xin + ((size_t)(unsigned)n << 7);
    float acc = b2[0];
    for (int j = 0; j < HDIM / 2; ++j) {
      float s = b1[j];
#pragma unroll 8
      for (int k = 0; k < HDIM; ++k) s += xr[k] * w1[k * (HDIM / 2) + j];
      acc += sspf(s) * w2[j];
    }
    atomicAdd(&out[batch[n]], acc);
  }
}

extern "C" void kernel_launch(void* const* d_in, const int* in_sizes, int n_in,
                              void* d_out, int out_size, void* d_ws, size_t ws_size,
                              hipStream_t stream) {
  const float* x     = (const float*)d_in[0];
  const float* pos   = (const float*)d_in[1];
  const int*   ei    = (const int*)d_in[2];
  const int*   batch = (const int*)d_in[3];
  const float* cf1   = (const float*)d_in[4];
  const float* nw1   = (const float*)d_in[5];
  const float* nb1   = (const float*)d_in[6];
  const float* nw2   = (const float*)d_in[7];
  const float* nb2   = (const float*)d_in[8];
  const float* cf2   = (const float*)d_in[9];
  const float* cb2   = (const float*)d_in[10];
  const float* pw    = (const float*)d_in[11];
  const float* pb    = (const float*)d_in[12];
  const float* h1w   = (const float*)d_in[13];
  const float* h1b   = (const float*)d_in[14];
  const float* h2w   = (const float*)d_in[15];
  const float* h2b   = (const float*)d_in[16];
  float* out = (float*)d_out;

  const int N = in_sizes[0] / HDIM;
  const int E = in_sizes[2] / 2;

  // workspace layout (256B-aligned slabs)
  char* base = (char*)d_ws;
  size_t off = 0;
  auto carve = [&](size_t bytes) -> char* {
    char* p = base + off;
    off = (off + bytes + 255) & ~(size_t)255;
    return p;
  };
  float* dbuf = (float*)carve((size_t)E * 4);
  float* ccut = (float*)carve((size_t)E * 4);
  float* xcur = (float*)carve((size_t)N * HDIM * 4);
  float* hbuf = (float*)carve((size_t)N * HDIM * 4);
  float* aggb = (float*)carve((size_t)N * HDIM * 4);
  _Float16* s_cf1[3]; _Float16* s_nw1[3]; _Float16* s_nw2[3];
  _Float16* s_cf2[3]; _Float16* s_pw[3];
  for (int l = 0; l < 3; ++l) {
    s_cf1[l] = (_Float16*)carve(32 * 512 * 2);
    s_nw1[l] = (_Float16*)carve(16 * 512 * 2);
    s_nw2[l] = (_Float16*)carve(32 * 512 * 2);
    s_cf2[l] = (_Float16*)carve(32 * 512 * 2);
    s_pw[l]  = (_Float16*)carve(32 * 512 * 2);
  }

  // one-time prep (deterministic; re-run every call)
  prep_edges<<<512, 256, 0, stream>>>(pos, ei, dbuf, ccut, E);
  for (int l = 0; l < 3; ++l) {
    swizzle_weight<<<64, 256, 0, stream>>>(cf1 + (size_t)l * 128 * 128, s_cf1[l], 128, 32);
    swizzle_weight<<<64, 256, 0, stream>>>(nw1 + (size_t)l * GCOUNT * 128, s_nw1[l], GCOUNT, 16);
    swizzle_weight<<<64, 256, 0, stream>>>(nw2 + (size_t)l * 128 * 128, s_nw2[l], 128, 32);
    swizzle_weight<<<64, 256, 0, stream>>>(cf2 + (size_t)l * 128 * 128, s_cf2[l], 128, 32);
    swizzle_weight<<<64, 256, 0, stream>>>(pw  + (size_t)l * 128 * 128, s_pw[l],  128, 32);
  }

  for (int l = 0; l < 3; ++l) {
    const float* xin = (l == 0) ? x : xcur;
    node_gemm<<<120, 256, 0, stream>>>(xin, s_cf1[l], hbuf, N);
    zerof<<<512, 256, 0, stream>>>(aggb, N * HDIM);
    edge_msg<<<600, 256, 0, stream>>>(dbuf, ccut, ei, hbuf, s_nw1[l], s_nw2[l],
                                      nb1 + (size_t)l * 128, nb2 + (size_t)l * 128,
                                      aggb, E);
    node_mlp2<<<120, 256, 0, stream>>>(aggb, s_cf2[l], cb2 + (size_t)l * 128,
                                       s_pw[l], pb + (size_t)l * 128, xcur, N);
  }

  zerof<<<1, 64, 0, stream>>>(out, out_size);
  head_reduce<<<240, 256, 0, stream>>>(xcur, h1w, h1b, h2w, h2b, batch, out, N);
}
